// HierarchicalMemory_64158221467656
// MI455X (gfx1250) — matrix operand order repaired and checked by reading, not compiled, not run on hardware
//
#include <hip/hip_runtime.h>
#include <hip/hip_bf16.h>
#include <math.h>

// ---------------------------------------------------------------------------
// Problem constants (match reference: B=256, N=100000, D=1024, k=16)
// ---------------------------------------------------------------------------
#define BQ        256        // queries
#define NMEM      100000     // memory rows
#define DIMS      1024       // feature dim
#define TOPK      16
#define EPSV      1e-8f

#define NQB       64         // queries per y-block (4 blocks)
#define NYBLK     (BQ / NQB)
#define WAVES     8          // 256 threads, wave32
#define TILE_COLS 128        // 8 waves x 16 cols
#define TILES_PER_SPLIT 8
#define SPLIT_COLS (TILE_COLS * TILES_PER_SPLIT)   // 1024
#define NSPLIT    98         // 98*1024 = 100352 >= 100000
#define KCH       32         // K per f16 WMMA
#define NCHUNK    (DIMS / KCH)
#define NBUF      4          // async LDS buffers (prefetch distance 2)

typedef __attribute__((ext_vector_type(16))) _Float16 v16h;
typedef __attribute__((ext_vector_type(8)))  _Float16 v8h;
typedef __attribute__((ext_vector_type(8)))  float    v8f;

// f32 -> f16 bits (RNE via hardware convert)
__device__ __forceinline__ unsigned short f2h(float f) {
    union { _Float16 h; unsigned short u; } x;
    x.h = (_Float16)f;
    return x.u;
}

// ---------------------------------------------------------------------------
// Kernel 1: convert f32 rows -> f16 and compute L2 norms.
// Block = one row (256 threads * float4 = 1024 elems). First 256 blocks are
// query rows, remaining 100000 are memory rows.
// ---------------------------------------------------------------------------
__global__ void hm_prep_kernel(const float* __restrict__ query,
                               const float* __restrict__ memory,
                               unsigned short* __restrict__ qh,
                               unsigned short* __restrict__ mh,
                               float* __restrict__ qn,
                               float* __restrict__ mn) {
    __shared__ float red[256];
    int row = blockIdx.x;
    const float* src; unsigned short* dst; float* nrm; int r;
    if (row < BQ) { src = query;  dst = qh; nrm = qn; r = row; }
    else          { src = memory; dst = mh; nrm = mn; r = row - BQ; }

    int t = threadIdx.x;
    const float4 v = *(const float4*)(src + (size_t)r * DIMS + (size_t)t * 4);
    float s = v.x * v.x + v.y * v.y + v.z * v.z + v.w * v.w;

    union { unsigned short h[4]; unsigned long long ll; } p;
    p.h[0] = f2h(v.x); p.h[1] = f2h(v.y); p.h[2] = f2h(v.z); p.h[3] = f2h(v.w);
    *(unsigned long long*)(dst + (size_t)r * DIMS + (size_t)t * 4) = p.ll;

    red[t] = s;
    __syncthreads();
    for (int off = 128; off > 0; off >>= 1) {
        if (t < off) red[t] += red[t + off];
        __syncthreads();
    }
    if (t == 0) nrm[r] = sqrtf(red[0]);
}

// ---------------------------------------------------------------------------
// Kernel 2: f16 WMMA GEMM (sims = Q * M^T / max(|q||m|,eps)) fused with a
// streaming per-query top-16 over this block's 1024-column split.
// grid = (NSPLIT, NYBLK), block = 256 (8 waves).
//   wave w owns 16 columns of each 128-col tile; per K-chunk its B tile is
//   DMA'd into its private LDS slice with global_load_async_to_lds_b128
//   (4 buffers, prefetch distance 2, s_wait_asynccnt); A fragments from L2.
// ---------------------------------------------------------------------------
__global__ void hm_gemm_topk_kernel(const unsigned short* __restrict__ mh,
                                    const unsigned short* __restrict__ qh,
                                    const float* __restrict__ qn,
                                    const float* __restrict__ mn,
                                    float* __restrict__ cand_val,
                                    int* __restrict__ cand_idx) {
    // Per-wave quad-buffered B tiles: [buf][wave][col(16)][K(32)] f16 = 32 KB
    __shared__ unsigned short sB[NBUF][WAVES][16][KCH];
    __shared__ float simbuf[NQB][TILE_COLS + 1];     // 33 KB, +1 anti-bank-conflict
    __shared__ float tv[NQB][TOPK];                  // running top values (desc)
    __shared__ int   ti[NQB][TOPK];                  // running top indices

    const int tid   = threadIdx.x;
    const int w     = tid >> 5;          // wave id 0..7
    const int lane  = tid & 31;
    const int lhalf = (lane < 16) ? 0 : 1;
    const int l16   = lane & 15;
    const int qbase = blockIdx.y * NQB;
    const long long splitbase = (long long)blockIdx.x * SPLIT_COLS;
    const unsigned long long mbase = (unsigned long long)(size_t)mh;

    // init running top-k lists
    for (int i = tid; i < NQB * TOPK; i += 256) {
        ((float*)tv)[i] = -INFINITY;
        ((int*)ti)[i]   = 0;
    }
    __syncthreads();

    for (int tile = 0; tile < TILES_PER_SPLIT; ++tile) {
        const long long cb = splitbase + (long long)tile * TILE_COLS;
        const int mycol = (int)cb + w * 16 + l16;                  // B column = memory row
        const int browc = (mycol < NMEM) ? mycol : (NMEM - 1);     // clamped for loads

        // async DMA of one 16x32 f16 B chunk into this wave's LDS slice:
        // per lane 32 contiguous bytes (2 x b128); B layout used here (contiguous k):
        //   lanes 0-15 hold K 0..15, lanes 16-31 hold K 16..31 of column l16.
        auto issueB = [&](int kc) {
            const int buf = kc & (NBUF - 1);
            unsigned ldsa = (unsigned)(size_t)(&sB[buf][w][l16][lhalf * 16]);
            unsigned voff = (unsigned)browc * (DIMS * 2u)
                          + (unsigned)kc * (KCH * 2u)
                          + (unsigned)lhalf * 32u;
            asm volatile("global_load_async_to_lds_b128 %0, %1, %2"
                         :: "v"(ldsa), "v"(voff), "s"(mbase) : "memory");
            asm volatile("global_load_async_to_lds_b128 %0, %1, %2 offset:16"
                         :: "v"(ldsa), "v"(voff), "s"(mbase) : "memory");
        };

        v8f acc[4] = {v8f{}, v8f{}, v8f{}, v8f{}};

        // All four A fragments first (one wait), then 4 back-to-back WMMAs.
        // 16-bit A layout: lanes 0-15: K 0..15; lanes 16-31: K 16..31 (matches B).
        auto computeChunk = [&](int kc) {
            const int buf = kc & (NBUF - 1);
            const v16h bfrag = *(const v16h*)&sB[buf][w][l16][lhalf * 16];
            v8h alo[4], ahi[4];
#pragma unroll
            for (int t = 0; t < 4; ++t) {
                const unsigned short* ap = qh
                    + (size_t)(qbase + 16 * t + l16) * DIMS
                    + (size_t)kc * KCH + (size_t)lhalf * 16;
                alo[t] = *(const v8h*)ap;          // K 0..7   (or 16..23)
                ahi[t] = *(const v8h*)(ap + 8);    // K 8..15  (or 24..31)
            }
#pragma unroll
            for (int t = 0; t < 4; ++t) {
                v16h a = __builtin_shufflevector(alo[t], ahi[t],
                    0, 1, 2, 3, 4, 5, 6, 7, 8, 9, 10, 11, 12, 13, 14, 15);
                acc[t] = __builtin_amdgcn_wmma_f32_16x16x32_f16(
                    false, a, false, bfrag, (short)0, acc[t], false, false);
            }
        };

        // Software pipeline: prefetch distance 2, branch-free steady state.
        issueB(0);
        issueB(1);
        for (int kc = 0; kc < NCHUNK - 2; ++kc) {
            issueB(kc + 2);
            asm volatile("s_wait_asynccnt %0" :: "i"(4) : "memory");
            computeChunk(kc);
        }
        asm volatile("s_wait_asynccnt %0" :: "i"(2) : "memory");
        computeChunk(NCHUNK - 2);
        asm volatile("s_wait_asynccnt %0" :: "i"(0) : "memory");
        computeChunk(NCHUNK - 1);

        // Scale to cosine sims and dump tile to LDS.
        // C layout: VGPR r -> row 16t + r (+8 for lanes 16-31), col = lane%16.
        const bool  colvalid = (mycol < NMEM);
        const float mnv = mn[browc];
#pragma unroll
        for (int t = 0; t < 4; ++t) {
#pragma unroll
            for (int r = 0; r < 8; ++r) {
                const int qloc = 16 * t + r + lhalf * 8;
                const float qnv = qn[qbase + qloc];
                const float sim = acc[t][r] / fmaxf(qnv * mnv, EPSV);
                simbuf[qloc][w * 16 + l16] = colvalid ? sim : -INFINITY;
            }
        }
        __syncthreads();

        // Streaming top-16 update: one thread per query row, threshold test
        // against current 16th-best (passes become rare quickly).
        if (tid < NQB) {
            const int q = tid;
            float vmin = tv[q][TOPK - 1];
            for (int c = 0; c < TILE_COLS; ++c) {
                const float v = simbuf[q][c];
                if (v > vmin) {
                    int pos = TOPK - 1;
                    while (pos > 0 && tv[q][pos - 1] < v) {
                        tv[q][pos] = tv[q][pos - 1];
                        ti[q][pos] = ti[q][pos - 1];
                        --pos;
                    }
                    tv[q][pos] = v;
                    ti[q][pos] = (int)(cb + c);
                    vmin = tv[q][TOPK - 1];
                }
            }
        }
        __syncthreads();
    }

    // Emit this split's 16 candidates per query.
    if (tid < NQB) {
        const int q = qbase + tid;
        const size_t base = ((size_t)q * NSPLIT + blockIdx.x) * TOPK;
#pragma unroll
        for (int j = 0; j < TOPK; ++j) {
            cand_val[base + j] = tv[tid][j];
            cand_idx[base + j] = ti[tid][j];
        }
    }
}

// ---------------------------------------------------------------------------
// Kernel 3: merge 98 sorted 16-candidate lists per query -> final top-16.
// Tiny serial pass (98*16 thresholded compares per query).
// ---------------------------------------------------------------------------
__global__ void hm_merge_kernel(const float* __restrict__ cand_val,
                                const int* __restrict__ cand_idx,
                                float* __restrict__ out_val,
                                int* __restrict__ out_idx) {
    __shared__ float lv[TOPK];
    __shared__ int   li[TOPK];
    const int q = blockIdx.x;
    if (threadIdx.x == 0) {
        for (int j = 0; j < TOPK; ++j) { lv[j] = -INFINITY; li[j] = 0; }
        for (int s = 0; s < NSPLIT; ++s) {
            const float* cv = cand_val + ((size_t)q * NSPLIT + s) * TOPK;
            const int*   ci = cand_idx + ((size_t)q * NSPLIT + s) * TOPK;
            for (int j = 0; j < TOPK; ++j) {
                const float v = cv[j];
                if (v <= lv[TOPK - 1]) break;   // per-split list is descending
                int pos = TOPK - 1;
                while (pos > 0 && lv[pos - 1] < v) {
                    lv[pos] = lv[pos - 1]; li[pos] = li[pos - 1]; --pos;
                }
                lv[pos] = v; li[pos] = ci[j];
            }
        }
        for (int j = 0; j < TOPK; ++j) {
            out_val[(size_t)q * TOPK + j] = lv[j];
            out_idx[(size_t)q * TOPK + j] = li[j];
        }
    }
}

// ---------------------------------------------------------------------------
// Host launcher. Workspace layout (bytes, 256-aligned blocks):
//   [0,            204800000)  memory f16    (100000 x 1024 x 2)
//   [204800000,    205324288)  query  f16    (256 x 1024 x 2)
//   [205324288,    205325312)  query norms   (256 f32)
//   [205325312,    205725440)  memory norms  (100000 f32, padded)
//   [205725440,    207331072)  cand_val      (256 x 98 x 16 f32)
//   [207331072,    208936704)  cand_idx      (256 x 98 x 16 i32)
// ---------------------------------------------------------------------------
extern "C" void kernel_launch(void* const* d_in, const int* in_sizes, int n_in,
                              void* d_out, int out_size, void* d_ws, size_t ws_size,
                              hipStream_t stream) {
    const float* query  = (const float*)d_in[0];
    const float* memory = (const float*)d_in[1];

    char* ws = (char*)d_ws;
    unsigned short* mh = (unsigned short*)(ws);
    unsigned short* qh = (unsigned short*)(ws + 204800000ull);
    float* qn          = (float*)(ws + 205324288ull);
    float* mn          = (float*)(ws + 205325312ull);
    float* cand_val    = (float*)(ws + 205725440ull);
    int*   cand_idx    = (int*)  (ws + 207331072ull);

    float* out_val = (float*)d_out;                 // [256,16] values
    int*   out_idx = (int*)d_out + BQ * TOPK;       // [256,16] indices (int32 bits)

    hm_prep_kernel<<<BQ + NMEM, 256, 0, stream>>>(query, memory, qh, mh, qn, mn);
    hm_gemm_topk_kernel<<<dim3(NSPLIT, NYBLK), 256, 0, stream>>>(
        mh, qh, qn, mn, cand_val, cand_idx);
    hm_merge_kernel<<<BQ, 32, 0, stream>>>(cand_val, cand_idx, out_val, out_idx);
}
